// SAFusion_52913997086889
// MI455X (gfx1250) — matrix operand order repaired and checked by reading, not compiled, not run on hardware
//
#include <hip/hip_runtime.h>

// ---------------------------------------------------------------------------
// Problem constants (from reference): B=4, NG=NP=2048 -> N=4096, D=512
// ---------------------------------------------------------------------------
#define B_    4
#define NG_   2048
#define N_    4096
#define D_    512
#define KB    32      // keys per attention inner block
#define KSTR  520     // padded LDS row stride (bf16 elems): 512 + 8 (TDM pad)

typedef __bf16 bf16_t;
typedef __attribute__((ext_vector_type(16))) __bf16 v16bf;
typedef __attribute__((ext_vector_type(8)))  float  v8f;
typedef __attribute__((ext_vector_type(4)))  unsigned int v4u;
typedef __attribute__((ext_vector_type(8)))  int v8i;
typedef __attribute__((ext_vector_type(4)))  int v4i;

// LDS layout (dynamic shared) for the attention kernel
#define KBUF_BYTES   (2 * KB * KSTR * 2)            // 66,560: double-buffered keys
#define SPART_OFF    (KBUF_BYTES)                   // 8 waves x 512 f32 = 16,384
#define PLDS_OFF     (SPART_OFF + 8 * 512 * 4)      // 2 x 512 bf16 = 2,048
#define ROWSUM_OFF   (PLDS_OFF + 2 * 512 * 2)       // 32 f32 = 128
#define SMEM_BYTES   (ROWSUM_OFF + 32 * 4)          // 85,120 total

// ---------------------------------------------------------------------------
// WMMA fragment loaders (16-bit data, wave32; A and B share this file's contiguous k order)
// A matrix 16(M)x32(K): lane L (g=L/16, m=L%16): elem j -> K = g*16 + j
// B matrix 32(K)x16(N): lane L (g=L/16, n=L%16): elem j -> K = g*16 + j
// C/D 16x16 f32: VGPR r, lane L: M = r + 8*(L/16), N = L%16
// (These per-lane-contiguous gathers collapse into ds_load_b128 pairs.)
// ---------------------------------------------------------------------------
__device__ inline v16bf load_a_frag(const bf16_t* src, int ld, int lane) {
    int g = lane >> 4, m = lane & 15;
    v16bf a;
#pragma unroll
    for (int j = 0; j < 16; ++j) {
        int k = g * 16 + j;
        a[j] = src[m * ld + k];
    }
    return a;
}

// B fragment where memory is laid out [n][k] -- used for S = Q * K^T.
__device__ inline v16bf load_bT_frag(const bf16_t* src, int ld, int lane) {
    int g = lane >> 4, n = lane & 15;
    v16bf b;
#pragma unroll
    for (int j = 0; j < 16; ++j) {
        int k = g * 16 + j;
        b[j] = src[n * ld + k];
    }
    return b;
}

// B fragment where memory is laid out [k][n] -- used for O += P * V.
__device__ inline v16bf load_b_frag(const bf16_t* src, int ld, int lane) {
    int g = lane >> 4, n = lane & 15;
    v16bf b;
#pragma unroll
    for (int j = 0; j < 16; ++j) {
        int k = g * 16 + j;
        b[j] = src[k * ld + n];
    }
    return b;
}

// ---------------------------------------------------------------------------
// TDM: issue an async DMA of one 32x512 bf16 key block (2-D tile) into LDS,
// with descriptor-driven padding of 4 DWORDs every 256 DWORDs so that rows
// land at the KSTR=520 stride.  Tracked by TENSORcnt.
// D# layout per CDNA5 ISA ch.8 (group0: flags/lds/global addr; group1: dims).
// ---------------------------------------------------------------------------
__device__ inline void tdm_load_kblock(const bf16_t* gsrc, unsigned int lds_off) {
    unsigned long long ga = (unsigned long long)(uintptr_t)gsrc;
    v4u g0;
    g0[0] = 1u;                                              // count=1 valid descriptor
    g0[1] = lds_off;                                         // lds_addr (bytes)
    g0[2] = (unsigned int)ga;                                // global_addr[31:0]
    g0[3] = (unsigned int)((ga >> 32) & 0x01FFFFFFu)         // global_addr[56:32]
          | 0x80000000u;                                     // type=2 ("image")
    v8i g1;
    g1[0] = (int)((1u << 16)      // data_size = 1 -> 2-byte elements
                | (1u << 20)      // pad_enable
                | (7u << 22)      // pad_interval = 7 -> every 256 DWORDs (=512 bf16)
                | (3u << 25));    // pad_amount  = 3 -> 4 DWORDs (=8 bf16)
    g1[1] = (int)(512u << 16);                 // tensor_dim0 = 512 (bits 79:48)
    g1[2] = (int)((unsigned)N_ << 16);         // tensor_dim0 hi | tensor_dim1 lo
    g1[3] = (int)(512u << 16);                 // tensor_dim1 hi | tile_dim0 = 512
    g1[4] = (int)KB;                           // tile_dim1 = 32, tile_dim2 = 0
    g1[5] = 512;                               // tensor_dim0_stride = 512 elements
    g1[6] = 0;                                 // stride hi | tensor_dim1_stride lo
    g1[7] = 0;
    v4i gz = {0, 0, 0, 0};
#if defined(__clang_major__) && (__clang_major__ >= 23)
    v8i gz8 = {0, 0, 0, 0, 0, 0, 0, 0};
    __builtin_amdgcn_tensor_load_to_lds(g0, g1, gz, gz, gz8, 0);
#else
    __builtin_amdgcn_tensor_load_to_lds(g0, g1, gz, gz, 0);
#endif
}

// ---------------------------------------------------------------------------
// Kernel 1: s = x / ||x||  (expmap0 + hyp_normalize collapse), store bf16
// ---------------------------------------------------------------------------
__global__ __launch_bounds__(128) void normalize_kernel(const float* __restrict__ g,
                                                        const float* __restrict__ p,
                                                        bf16_t* __restrict__ Sbf) {
    int token = blockIdx.x;
    int b = token / N_;
    int n = token % N_;
    const float* src = (n < NG_) ? (g + ((size_t)b * NG_ + n) * D_)
                                 : (p + ((size_t)b * (N_ - NG_) + (n - NG_)) * D_);
    int tid = threadIdx.x;
    float x[4];
    float ss = 0.f;
#pragma unroll
    for (int i = 0; i < 4; ++i) { x[i] = src[tid + i * 128]; ss += x[i] * x[i]; }
#pragma unroll
    for (int m = 16; m > 0; m >>= 1) ss += __shfl_xor(ss, m, 32);
    __shared__ float sred[4];
    if ((tid & 31) == 0) sred[tid >> 5] = ss;
    __syncthreads();
    float tot = sred[0] + sred[1] + sred[2] + sred[3];
    float rn = rsqrtf(fmaxf(tot, 1e-7f));
    bf16_t* dst = Sbf + (size_t)token * D_;
#pragma unroll
    for (int i = 0; i < 4; ++i) dst[tid + i * 128] = (bf16_t)(x[i] * rn);
}

// ---------------------------------------------------------------------------
// Kernel 2: attention.  scores = (2 - s_n . s_m)/scale + bias, softmax, W @ S.
// 256 threads = 8 waves; 32 queries/block; wave (qt, dc):
//   qt = wave&1 -> 16-query sub-tile, dc = wave>>1 -> 128-dim feature chunk.
// Keys double-buffered in LDS via the Tensor Data Mover (TENSORcnt-gated).
// B-fragment loads are batched ahead of each WMMA chain so multiple
// ds_load_b128s stay in flight under the matrix pipe.
// ---------------------------------------------------------------------------
__global__ __launch_bounds__(256) void attn_kernel(const bf16_t* __restrict__ Sbf,
                                                   const float* __restrict__ scale_p,
                                                   const float* __restrict__ bias_p,
                                                   float* __restrict__ AVE) {
    extern __shared__ __align__(16) char smem[];
    bf16_t* Kbuf   = (bf16_t*)smem;                   // [2][KB*KSTR]
    float*  Spart  = (float*)(smem + SPART_OFF);      // [8][512]
    bf16_t* Plds   = (bf16_t*)(smem + PLDS_OFF);      // [2][512]
    float*  rowsum = (float*)(smem + ROWSUM_OFF);     // [32]

    const int tid  = threadIdx.x;
    const int lane = tid & 31;
    const int wv   = tid >> 5;
    const int qt   = wv & 1;
    const int dc   = wv >> 1;
    const int gh   = lane >> 4;
    const int nl   = lane & 15;

    const int b  = blockIdx.x >> 7;          // 128 blocks per batch
    const int q0 = (blockIdx.x & 127) * 32;

    const float invscale = 1.0f / scale_p[0];
    const float biasv    = bias_p[0];

    if (tid < 32) rowsum[tid] = 0.0f;

    const unsigned int kbuf_lds_base = (unsigned int)(uintptr_t)(void*)Kbuf;
    const bf16_t* kbase_g = Sbf + (size_t)b * N_ * D_;

    // Q fragments for this wave: 16 queries x 128 dims = 4 A-fragments (K=32)
    v16bf qf[4];
#pragma unroll
    for (int f = 0; f < 4; ++f) {
        const bf16_t* qsrc = Sbf + ((size_t)b * N_ + q0 + qt * 16) * D_ + dc * 128 + f * 32;
        qf[f] = load_a_frag(qsrc, D_, lane);
    }

    v8f oacc[8] = {};        // 16 queries x 128 dims f32 accumulator
    float lsum = 0.0f;       // per-lane partial softmax denominator

    // prime the pipeline: DMA key block 0 into buffer 0 (wave 0 only; TDM
    // issue is EXEC-independent, so a uniform wave guard gives one issue)
    if (wv == 0) {
        tdm_load_kblock(kbase_g, kbuf_lds_base);
    }

    const int nblocks = N_ / KB;
    for (int kb = 0; kb < nblocks; ++kb) {
        if (wv == 0) { __builtin_amdgcn_s_wait_tensorcnt(0); }
        __syncthreads();  // buffer (kb&1) DMA complete; prev iter fully consumed

        // kick off the DMA for the next key block into the other buffer while
        // this block is being consumed
        if (wv == 0 && (kb + 1) < nblocks) {
            tdm_load_kblock(kbase_g + (size_t)(kb + 1) * KB * D_,
                            kbuf_lds_base + (unsigned int)(((kb + 1) & 1) * KB * KSTR * 2));
        }

        const bf16_t* Kbf = Kbuf + (size_t)(kb & 1) * KB * KSTR;

        // ---- partial S = Q_slice * K_slice^T (16x32 tile over 128-dim slice) ----
        // Batch all 4 B-fragment loads ahead of the WMMA chain.
#pragma unroll
        for (int t = 0; t < 2; ++t) {
            v16bf kfr[4];
#pragma unroll
            for (int f = 0; f < 4; ++f) {
                kfr[f] = load_bT_frag(&Kbf[(t * 16) * KSTR + dc * 128 + f * 32], KSTR, lane);
            }
            v8f c = {};
#pragma unroll
            for (int f = 0; f < 4; ++f) {
                c = __builtin_amdgcn_wmma_f32_16x16x32_bf16(false, qf[f], false, kfr[f],
                                                            (short)0, c, false, false);
            }
#pragma unroll
            for (int r = 0; r < 8; ++r) {
                int M = r + 8 * gh;
                Spart[wv * 512 + M * 32 + t * 16 + nl] = c[r];
            }
        }
        __syncthreads();

        // ---- reduce partials across the 4 D-chunk waves, exp, store P ----
        {
            int base = dc * 128 + lane * 4;
#pragma unroll
            for (int i = 0; i < 4; ++i) {
                int idx = base + i;
                float s4 = Spart[qt * 512 + idx] + Spart[(qt + 2) * 512 + idx] +
                           Spart[(qt + 4) * 512 + idx] + Spart[(qt + 6) * 512 + idx];
                float pr = __expf((2.0f - s4) * invscale + biasv);
                Plds[qt * 512 + idx] = (bf16_t)pr;
                lsum += pr;
            }
        }
        __syncthreads();

        // ---- O += P * V on this wave's 128-dim output slice ----
        // V-fragments batched in groups of 4; the 4 WMMAs per group are
        // independent (distinct accumulators) and can issue back-to-back.
        v16bf pf = load_a_frag(&Plds[qt * 512], 32, lane);
#pragma unroll
        for (int oh = 0; oh < 2; ++oh) {
            v16bf vf[4];
#pragma unroll
            for (int i = 0; i < 4; ++i) {
                vf[i] = load_b_frag(&Kbf[dc * 128 + (oh * 4 + i) * 16], KSTR, lane);
            }
#pragma unroll
            for (int i = 0; i < 4; ++i) {
                oacc[oh * 4 + i] = __builtin_amdgcn_wmma_f32_16x16x32_bf16(
                    false, pf, false, vf[i], (short)0, oacc[oh * 4 + i], false, false);
            }
        }
    }

    // ---- finish softmax denominators: lanes sharing a row reduce + ds_add ----
#pragma unroll
    for (int m = 1; m <= 4; m <<= 1) lsum += __shfl_xor(lsum, m, 32);
    if ((lane & 7) == 0) atomicAdd(&rowsum[qt * 16 + dc * 4 + (lane >> 3)], lsum);
    __syncthreads();

    // ---- divide by row sums and write AVE ----
    float* outbase = AVE + ((size_t)b * N_ + q0 + qt * 16) * D_ + dc * 128;
#pragma unroll
    for (int ot = 0; ot < 8; ++ot) {
#pragma unroll
        for (int r = 0; r < 8; ++r) {
            int M = r + 8 * gh;
            float ell = rowsum[qt * 16 + M];
            outbase[(size_t)M * D_ + ot * 16 + nl] = oacc[ot][r] / ell;
        }
    }
}

// ---------------------------------------------------------------------------
// Kernel 3: epilogue -- hyp_coattn denom, mobius_add, hyp_layernorm, logmap0.
// One token per block, 256 threads, 2 dims per thread.
// ---------------------------------------------------------------------------
__device__ inline float block_sum256(float v, volatile float* sh) {
#pragma unroll
    for (int m = 16; m > 0; m >>= 1) v += __shfl_xor(v, m, 32);
    __syncthreads();
    if ((threadIdx.x & 31) == 0) sh[threadIdx.x >> 5] = v;
    __syncthreads();
    return sh[0] + sh[1] + sh[2] + sh[3] + sh[4] + sh[5] + sh[6] + sh[7];
}

__global__ __launch_bounds__(256) void epilogue_kernel(const bf16_t* __restrict__ Sbf,
                                                       const float* __restrict__ AVE,
                                                       const float* __restrict__ gamma,
                                                       const float* __restrict__ beta,
                                                       float* __restrict__ out) {
    __shared__ float sred[8];
    const int token = blockIdx.x;
    const int tid = threadIdx.x;
    const float SQRT2 = 1.41421356237309515f;

    const bf16_t* sp = Sbf + (size_t)token * D_;
    const float*  av = AVE + (size_t)token * D_;
    float s0 = (float)sp[tid], s1 = (float)sp[tid + 256];
    float a0 = av[tid],        a1 = av[tid + 256];

    // hyp_coattn tail: denom = sqrt(|l_inner(ave,ave)|), ave_time = sqrt(2)
    float r1 = block_sum256(a0 * a0 + a1 * a1, sred);   // ||ave_space||^2
    float r2 = block_sum256(s0 * a0 + s1 * a1, sred);   // s . ave_space
    float denom = sqrtf(fmaxf(fabsf(r1 - 2.0f), 1e-7f));

    // mobius_add(xh=[sqrt2,s], y=ave/denom)
    float yt    = SQRT2 / denom;
    float alpha = fmaxf(yt, 1.0f + 1e-7f);
    float coefu = acoshf(alpha) * rsqrtf(alpha * alpha - 1.0f);
    float cu    = coefu / denom;                // u_space = cu * ave_space
    float dot_su = cu * r2;                     // l_inner(x,u) = s.u_space
    float coef   = dot_su / (1.0f + SQRT2);
    float vt = coef * (SQRT2 + 1.0f);
    float v0 = cu * a0 + coef * s0;
    float v1 = cu * a1 + coef * s1;
    float r3 = block_sum256(v0 * v0 + v1 * v1, sred);
    float nv = sqrtf(fmaxf(r3 - vt * vt, 1e-7f));
    float ch = coshf(nv), sh = sinhf(nv) / nv;
    float z0 = ch * s0 + sh * v0;
    float z1 = ch * s1 + sh * v1;

    // hyp_layernorm on space part
    float mu  = block_sum256(z0 + z1, sred) * (1.0f / D_);
    float zc0 = z0 - mu, zc1 = z1 - mu;
    float var = block_sum256(zc0 * zc0 + zc1 * zc1, sred) * (1.0f / D_);
    float rstd = rsqrtf(var + 1e-5f);
    float zn0 = zc0 * rstd * gamma[tid]       + beta[tid];
    float zn1 = zc1 * rstd * gamma[tid + 256] + beta[tid + 256];

    // logmap0 (drop time component; output is space part only)
    float r6 = block_sum256(zn0 * zn0 + zn1 * zn1, sred);
    float t  = sqrtf(r6 + 1.0f);
    float a2 = fmaxf(t, 1.0f + 1e-7f);
    float co = acoshf(a2) * rsqrtf(a2 * a2 - 1.0f);

    float* op = out + (size_t)token * D_;
    op[tid]       = co * zn0;
    op[tid + 256] = co * zn1;
}

// ---------------------------------------------------------------------------
// Launch
// ---------------------------------------------------------------------------
extern "C" void kernel_launch(void* const* d_in, const int* in_sizes, int n_in,
                              void* d_out, int out_size, void* d_ws, size_t ws_size,
                              hipStream_t stream) {
    const float* g     = (const float*)d_in[0];
    const float* p     = (const float*)d_in[1];
    const float* scale = (const float*)d_in[2];
    const float* bias  = (const float*)d_in[3];
    const float* gamma = (const float*)d_in[4];
    const float* beta  = (const float*)d_in[5];
    float* out = (float*)d_out;

    // workspace layout: [ Sbf: B*N*D bf16 (16 MB) | AVE: B*N*D f32 (32 MB) ]
    bf16_t* Sbf = (bf16_t*)d_ws;
    float*  AVE = (float*)((char*)d_ws + (size_t)B_ * N_ * D_ * sizeof(bf16_t));

    normalize_kernel<<<B_ * N_, 128, 0, stream>>>(g, p, Sbf);
    attn_kernel<<<B_ * (N_ / 32), 256, SMEM_BYTES, stream>>>(Sbf, scale, bias, AVE);
    epilogue_kernel<<<B_ * N_, 256, 0, stream>>>(Sbf, AVE, gamma, beta, out);
}
